// EquivariantAttention_53721450939092
// MI455X (gfx1250) — matrix operand order repaired and checked by reading, not compiled, not run on hardware
//
#include <hip/hip_runtime.h>
#include <hip/hip_bf16.h>
#include <math.h>

// ---------------------------------------------------------------------------
// EquivariantAttention for MI455X (gfx1250, wave32, WMMA)
//   B=2 N=2048 DIM=256 H=4 DH=64 M=16 NN=32 FF=4 POS_DIM=9
// ---------------------------------------------------------------------------

typedef __attribute__((ext_vector_type(16))) _Float16 v16h;
typedef __attribute__((ext_vector_type(8)))  float    v8f;

#define NB   2
#define NN_  2048
#define BIH  11          // log2(N)
#define HH   4
#define DHH  64
#define MM_  16
#define KNN  32

// pack two contiguous float4 (8 floats) into v16h elements [e0..e0+7]
__device__ __forceinline__ void pack8(v16h& a, int e0, const float4& x, const float4& y) {
  a[e0 + 0] = (_Float16)x.x; a[e0 + 1] = (_Float16)x.y;
  a[e0 + 2] = (_Float16)x.z; a[e0 + 3] = (_Float16)x.w;
  a[e0 + 4] = (_Float16)y.x; a[e0 + 5] = (_Float16)y.y;
  a[e0 + 6] = (_Float16)y.z; a[e0 + 7] = (_Float16)y.w;
}

// ---------------------------------------------------------------------------
// Kernel 1: exact KNN (matches jax.lax.top_k(-dist): ascending dist, ties -> lower index)
// ---------------------------------------------------------------------------
__global__ __launch_bounds__(256) void knn_kernel(const float* __restrict__ coors,
                                                  int* __restrict__ g_idx,
                                                  float* __restrict__ g_dnb,
                                                  float* __restrict__ g_rel) {
  const int bi = blockIdx.x;            // b*N + i
  const int b  = bi >> BIH;
  const int t  = threadIdx.x;

  __shared__ float dist[NN_];
  __shared__ float rd[256];
  __shared__ int   ri[256];
  __shared__ float ci[3];

  if (t < 3) ci[t] = coors[(size_t)bi * 3 + t];
  __syncthreads();

  for (int j = t; j < NN_; j += 256) {
    const float* cj = coors + (size_t)(b * NN_ + j) * 3;
    float dx = ci[0] - cj[0], dy = ci[1] - cj[1], dz = ci[2] - cj[2];
    dist[j] = sqrtf(dx * dx + dy * dy + dz * dz);
  }
  __syncthreads();

  for (int s = 0; s < KNN; ++s) {
    float bd = 3.4e38f; int bj = 0x7fffffff;
    for (int j = t; j < NN_; j += 256) {
      float d = dist[j];
      if (d < bd || (d == bd && j < bj)) { bd = d; bj = j; }
    }
    rd[t] = bd; ri[t] = bj;
    __syncthreads();
    for (int off = 128; off > 0; off >>= 1) {
      if (t < off) {
        float od = rd[t + off]; int oj = ri[t + off];
        if (od < rd[t] || (od == rd[t] && oj < ri[t])) { rd[t] = od; ri[t] = oj; }
      }
      __syncthreads();
    }
    if (t == 0) {
      int j = ri[0];
      g_idx[bi * KNN + s] = j;
      g_dnb[bi * KNN + s] = rd[0];
      const float* cj = coors + (size_t)(b * NN_ + j) * 3;
      g_rel[(bi * KNN + s) * 3 + 0] = ci[0] - cj[0];
      g_rel[(bi * KNN + s) * 3 + 1] = ci[1] - cj[1];
      g_rel[(bi * KNN + s) * 3 + 2] = ci[2] - cj[2];
      dist[j] = 3.4e38f;
    }
    __syncthreads();
  }
}

// ---------------------------------------------------------------------------
// Generic f32-in / f16-WMMA / f32-out GEMM:  D = A(MxK) * B(KxN) (+bias)
// One wave per 16x16 output tile; documented CDNA5 fragment layouts.
// A fragment loaded as float4 pairs (contiguous 8-float runs per lane).
// ---------------------------------------------------------------------------
__global__ __launch_bounds__(256) void gemm_f16_kernel(const float* __restrict__ A,
                                                       const float* __restrict__ Bm,
                                                       const float* __restrict__ bias,
                                                       float* __restrict__ D,
                                                       int K, int ldA, int ldB, int ldD,
                                                       int ntn /* N/16 */) {
  const int gid  = blockIdx.x * 8 + (threadIdx.x >> 5);
  const int lane = threadIdx.x & 31;
  const int rt   = gid / ntn, ct = gid % ntn;
  const int row0 = rt * 16, col0 = ct * 16;
  const int m = lane & 15, hl = lane >> 4;
  const int coln = col0 + m;

  const float* arow = A + (size_t)(row0 + m) * ldA + 8 * hl;
  const float  bv   = bias ? bias[coln] : 0.0f;     // uniform, hoisted

  v8f acc = {};
  for (int k0 = 0; k0 < K; k0 += 32) {
    v16h a, bfr;
    // A: elements 0..7 -> K = k0+8*hl+[0,8), elements 8..15 -> K = k0+16+8*hl+[0,8)
    float4 x0 = *(const float4*)(arow + k0);
    float4 x1 = *(const float4*)(arow + k0 + 4);
    float4 y0 = *(const float4*)(arow + k0 + 16);
    float4 y1 = *(const float4*)(arow + k0 + 20);
    pack8(a, 0, x0, x1);
    pack8(a, 8, y0, y1);
    // B: elements 0..7 -> K = k0+8*hl+e, elements 8..15 -> K = k0+16+8*hl+(e-8) (column-strided)
#pragma unroll
    for (int e = 0; e < 16; ++e)
      bfr[e] = (_Float16)Bm[(size_t)(k0 + ((e < 8) ? (8 * hl + e) : (8 + 8 * hl + e))) * ldB + coln];
    acc = __builtin_amdgcn_wmma_f32_16x16x32_f16(false, a, false, bfr, (short)0, acc,
                                                 false, false);
  }
#pragma unroll
  for (int r = 0; r < 8; ++r) {
    int row = row0 + r + 8 * hl;
    D[(size_t)row * ldD + coln] = acc[r] + bv;
  }
}

// ---------------------------------------------------------------------------
// Kernel 3: fused per-(b,i) block — pos MLP, edge MLP (WMMA), attn + coor heads
// ---------------------------------------------------------------------------
__global__ __launch_bounds__(256) void fused_kernel(
    const float* __restrict__ qkvbuf,
    const int* __restrict__ g_idx, const float* __restrict__ g_dnb,
    const float* __restrict__ g_rel,
    const float* __restrict__ wp1, const float* __restrict__ bp1,
    const float* __restrict__ wp2, const float* __restrict__ bp2,
    const float* __restrict__ we1, const float* __restrict__ be1,
    const float* __restrict__ we2, const float* __restrict__ be2,
    const float* __restrict__ wa1, const float* __restrict__ ba1,
    const float* __restrict__ wa2, const float* __restrict__ ba2,
    const float* __restrict__ wc1, const float* __restrict__ bc1,
    const float* __restrict__ wc2, const float* __restrict__ bc2,
    float* __restrict__ attn_out, float* __restrict__ coors_out) {

  const int bi = blockIdx.x;        // == b*N + i (flat qkv row)
  const int b  = bi >> BIH;
  const int t  = threadIdx.x;

  __shared__ int      s_idx[KNN];
  __shared__ float    s_dnb[KNN];
  __shared__ float    s_rel[KNN][3];
  __shared__ float    s_q[HH][DHH];
  __shared__ float    s_pos[KNN][DHH];
  __shared__ float    s_ovl[4096];          // overlay: pos hidden (32x128) then WMMA staging
  __shared__ _Float16 s_we1[64 * 128];
  __shared__ _Float16 s_we2[128 * 16];
  __shared__ _Float16 s_m[128 * 20];        // m_ij, ld=20
  __shared__ float    s_simraw[HH][KNN];
  __shared__ float    s_attn[HH][KNN];
  __shared__ float    s_cw[KNN];

  // ---- stage small per-node data + f16 edge weights ----
  if (t < KNN) {
    s_idx[t] = g_idx[bi * KNN + t];
    s_dnb[t] = g_dnb[bi * KNN + t];
    s_cw[t]  = bc2[0];
  }
  if (t < KNN * 3) s_rel[t / 3][t % 3] = g_rel[bi * KNN * 3 + t];
  { int h = t >> 6, d = t & 63; s_q[h][d] = qkvbuf[(size_t)bi * 768 + h * 64 + d]; }
  for (int e = t; e < 64 * 128; e += 256) s_we1[e] = (_Float16)we1[e];
  for (int e = t; e < 128 * 16; e += 256) s_we2[e] = (_Float16)we2[e];
  __syncthreads();

  // ---- pos MLP: fe(9) -> relu(128) -> 64, scalar, hidden in overlay ----
  for (int item = t; item < KNN * 128; item += 256) {
    int nb = item >> 7, j = item & 127;
    float d = s_dnb[nb];
    float acc = bp1[j];
#pragma unroll
    for (int f = 0; f < 4; ++f) {
      float x = d / (float)(1 << f);
      acc += sinf(x) * wp1[f * 128 + j];
      acc += cosf(x) * wp1[(4 + f) * 128 + j];
    }
    acc += d * wp1[8 * 128 + j];
    s_ovl[item] = acc > 0.f ? acc : 0.f;
  }
  __syncthreads();
  for (int item = t; item < KNN * DHH; item += 256) {
    int nb = item >> 6, dd = item & 63;
    float acc = bp2[dd];
    for (int j = 0; j < 128; ++j) acc += s_ovl[nb * 128 + j] * wp2[j * 64 + dd];
    s_pos[nb][dd] = acc;
  }
  __syncthreads();

  // ---- edge MLP via WMMA: rows = h*32+nb (128), 64 -> relu(128) -> relu(16) ----
  {
    const int w = t >> 5, lane = t & 31;
    const int m = lane & 15, hl = lane >> 4;
    const int h = w >> 1;
    const int r_row = 16 * w + m;           // this lane's A-matrix row (0..127)
    const int nb = r_row & 31;
    const int j  = s_idx[nb];
    const float* krow = qkvbuf + (size_t)(b * NN_ + j) * 768 + 256 + h * 64 + 8 * hl;
    __builtin_prefetch(krow, 0, 1);
    _Float16* stg = ((_Float16*)s_ovl) + w * (16 * 34);   // per-wave 16x32 staging, ld=34

    // vectorized gather of this lane's k-row chunks (32B-aligned)
    float4 k00 = *(const float4*)(krow + 0);
    float4 k01 = *(const float4*)(krow + 4);
    float4 k10 = *(const float4*)(krow + 16);
    float4 k11 = *(const float4*)(krow + 20);
    float4 k20 = *(const float4*)(krow + 32);
    float4 k21 = *(const float4*)(krow + 36);
    float4 k30 = *(const float4*)(krow + 48);
    float4 k31 = *(const float4*)(krow + 52);

    // A fragments for K=0..31 and K=32..63 of edge_in = q - k_nb + pos
    v16h a0, a1;
    {
      const float* kf = (const float*)&k00;   // k00,k01 contiguous in regs via array below
      float kl0[8] = { k00.x, k00.y, k00.z, k00.w, k01.x, k01.y, k01.z, k01.w };
      float kl1[8] = { k10.x, k10.y, k10.z, k10.w, k11.x, k11.y, k11.z, k11.w };
      float kl2[8] = { k20.x, k20.y, k20.z, k20.w, k21.x, k21.y, k21.z, k21.w };
      float kl3[8] = { k30.x, k30.y, k30.z, k30.w, k31.x, k31.y, k31.z, k31.w };
      (void)kf;
#pragma unroll
      for (int e = 0; e < 8; ++e) {
        int k = 8 * hl + e;
        a0[e]     = (_Float16)(s_q[h][k]      - kl0[e] + s_pos[nb][k]);
        a0[e + 8] = (_Float16)(s_q[h][k + 16] - kl1[e] + s_pos[nb][k + 16]);
        a1[e]     = (_Float16)(s_q[h][k + 32] - kl2[e] + s_pos[nb][k + 32]);
        a1[e + 8] = (_Float16)(s_q[h][k + 48] - kl3[e] + s_pos[nb][k + 48]);
      }
    }

    v8f mAcc = {};
    for (int cc = 0; cc < 4; ++cc) {        // layer-2 K-step = layer-1 col pair
#pragma unroll
      for (int c2 = 0; c2 < 2; ++c2) {
        int c = cc * 2 + c2;                // layer-1 col tile (16 cols)
        v16h b0, b1;
#pragma unroll
        for (int e = 0; e < 16; ++e) {
          int kb = (e < 8) ? (8 * hl + e) : (8 + 8 * hl + e);
          b0[e] = s_we1[kb * 128 + 16 * c + m];
          b1[e] = s_we1[(kb + 32) * 128 + 16 * c + m];
        }
        v8f acc = {};
        acc = __builtin_amdgcn_wmma_f32_16x16x32_f16(false, a0, false, b0, (short)0, acc,
                                                     false, false);
        acc = __builtin_amdgcn_wmma_f32_16x16x32_f16(false, a1, false, b1, (short)0, acc,
                                                     false, false);
        const float be = be1[16 * c + m];   // uniform per lane
#pragma unroll
        for (int r = 0; r < 8; ++r) {       // bias + relu -> staging (wave-local LDS)
          float v = acc[r] + be;
          stg[(r + 8 * hl) * 34 + c2 * 16 + m] = (_Float16)(v > 0.f ? v : 0.f);
        }
      }
      // layer 2: A from staging (K=32 chunk), B from w_e2 rows 32*cc..
      v16h a2, b2;
#pragma unroll
      for (int e = 0; e < 16; ++e) {
        int kl = (e < 8) ? (8 * hl + e) : (8 + 8 * hl + e);
        a2[e] = stg[m * 34 + kl];
        b2[e] = s_we2[(32 * cc + kl) * 16 + m];
      }
      mAcc = __builtin_amdgcn_wmma_f32_16x16x32_f16(false, a2, false, b2, (short)0, mAcc,
                                                    false, false);
    }
    const float be2v = be2[m];
#pragma unroll
    for (int r = 0; r < 8; ++r) {           // m_ij = relu(. + b_e2)
      int row = 16 * w + r + 8 * hl;
      float v = mAcc[r] + be2v;
      s_m[row * 20 + m] = (_Float16)(v > 0.f ? v : 0.f);
    }
  }
  __syncthreads();

  // ---- heads: sim (threads 0..127), coor weights (threads 128..255) ----
  if (t < 128) {
    int r = t, h = r >> 5, nb = r & 31;
    float mrow[16];
#pragma unroll
    for (int mmi = 0; mmi < 16; ++mmi) mrow[mmi] = (float)s_m[r * 20 + mmi];
    float sim = ba2[0];
    for (int j = 0; j < 64; ++j) {
      float hv = ba1[j];
#pragma unroll
      for (int mmi = 0; mmi < 16; ++mmi) hv += mrow[mmi] * wa1[mmi * 64 + j];
      hv = hv > 0.f ? hv : 0.f;
      sim += hv * wa2[j];
    }
    s_simraw[h][nb] = sim;
  } else {
    int tt = t - 128, nb = tt >> 2, qq = tt & 3;
    float ci[64];                            // coor_in[nb][h*16+mm] = m_ij[h][nb][mm]
#pragma unroll
    for (int c = 0; c < 64; ++c) ci[c] = (float)s_m[((c >> 4) * 32 + nb) * 20 + (c & 15)];
    float partial = 0.f;
    for (int j = qq * 16; j < qq * 16 + 16; ++j) {
      float hv = bc1[j];
      for (int c = 0; c < 64; ++c) hv += ci[c] * wc1[c * 64 + j];
      hv = hv > 0.f ? hv : 0.f;
      partial += hv * wc2[j];
    }
    atomicAdd(&s_cw[nb], partial);
  }
  __syncthreads();

  // ---- softmax per head + equivariant coordinate output ----
  if (t < HH) {
    float mx = -3.4e38f;
    for (int nb = 0; nb < KNN; ++nb) mx = fmaxf(mx, s_simraw[t][nb]);
    float sum = 0.f, ex[KNN];
    for (int nb = 0; nb < KNN; ++nb) { ex[nb] = expf(s_simraw[t][nb] - mx); sum += ex[nb]; }
    for (int nb = 0; nb < KNN; ++nb) s_attn[t][nb] = ex[nb] / sum;
  } else if (t < HH + 3) {
    int c = t - HH;
    float acc = 0.f;
    for (int nb = 0; nb < KNN; ++nb) acc += s_cw[nb] * s_rel[nb][c];
    coors_out[bi * 3 + c] = acc;
  }
  __syncthreads();

  // ---- out_h = sum_nb attn * (v_nb + pos) ----
  {
    int h = t >> 6, dd = t & 63;
    float acc = 0.f;
    for (int nb = 0; nb < KNN; ++nb) {
      int j = s_idx[nb];
      acc += s_attn[h][nb] *
             (qkvbuf[(size_t)(b * NN_ + j) * 768 + 512 + h * 64 + dd] + s_pos[nb][dd]);
    }
    attn_out[(size_t)bi * 256 + h * 64 + dd] = acc;
  }
}

// ---------------------------------------------------------------------------
// Launcher
// ---------------------------------------------------------------------------
extern "C" void kernel_launch(void* const* d_in, const int* in_sizes, int n_in,
                              void* d_out, int out_size, void* d_ws, size_t ws_size,
                              hipStream_t stream) {
  const float* feats = (const float*)d_in[0];
  const float* coors = (const float*)d_in[1];
  const float* w_qkv = (const float*)d_in[2];
  const float* w_out = (const float*)d_in[3];
  const float* b_out = (const float*)d_in[4];
  const float* w_p1  = (const float*)d_in[5];
  const float* b_p1  = (const float*)d_in[6];
  const float* w_p2  = (const float*)d_in[7];
  const float* b_p2  = (const float*)d_in[8];
  const float* w_e1  = (const float*)d_in[9];
  const float* b_e1  = (const float*)d_in[10];
  const float* w_e2  = (const float*)d_in[11];
  const float* b_e2  = (const float*)d_in[12];
  const float* w_a1  = (const float*)d_in[13];
  const float* b_a1  = (const float*)d_in[14];
  const float* w_a2  = (const float*)d_in[15];
  const float* b_a2  = (const float*)d_in[16];
  const float* w_c1  = (const float*)d_in[17];
  const float* b_c1  = (const float*)d_in[18];
  const float* w_c2  = (const float*)d_in[19];
  const float* b_c2  = (const float*)d_in[20];

  float* ws   = (float*)d_ws;
  int*   idx  = (int*)ws;                 // 131072 ints
  float* dnb  = ws + 131072;              // 131072 f
  float* rel  = ws + 262144;              // 393216 f
  float* qkv  = ws + 655360;              // 4096*768 = 3145728 f
  float* aout = ws + 3801088;             // 4096*256 = 1048576 f

  float* out  = (float*)d_out;            // (B,N,256)
  float* cout = out + (size_t)NB * NN_ * 256;   // (B,N,3)

  // 1) KNN (idx / d_nb / rel_nb)
  knn_kernel<<<NB * NN_, 256, 0, stream>>>(coors, idx, dnb, rel);

  // 2) qkv = feats @ w_qkv   (4096x256 @ 256x768), WMMA f16
  gemm_f16_kernel<<<(256 * 48) / 8, 256, 0, stream>>>(feats, w_qkv, nullptr, qkv,
                                                      256, 256, 768, 768, 48);

  // 3) fused pos MLP + edge MLP (WMMA) + heads + softmax + aggregation
  fused_kernel<<<NB * NN_, 256, 0, stream>>>(qkv, idx, dnb, rel,
                                             w_p1, b_p1, w_p2, b_p2,
                                             w_e1, b_e1, w_e2, b_e2,
                                             w_a1, b_a1, w_a2, b_a2,
                                             w_c1, b_c1, w_c2, b_c2,
                                             aout, cout);

  // 4) out = aout @ w_out + b_out   (4096x256 @ 256x256), WMMA f16
  gemm_f16_kernel<<<(256 * 16) / 8, 256, 0, stream>>>(aout, w_out, b_out, out,
                                                      256, 256, 256, 256, 16);
}
